// DenseTransformerLayer_17927193494088
// MI455X (gfx1250) — compile-verified, matrix operand order checked by reading, not run on hardware
//
#include <hip/hip_runtime.h>
#include <hip/hip_bf16.h>

// =====================================================================
// MI455X (gfx1250, wave32) dense graph-transformer layer.
// Matrix work: v_wmma_f32_16x16x32_bf16 (fp32 accumulate); fp32 tensors
// converted to bf16 while staging into LDS. Large bf16 operands are
// DMA'd into LDS with the Tensor Data Mover (tensor_load_to_lds +
// s_wait_tensorcnt), using D# padding to produce the bank-padded pitch.
//
// Scratch layout (requires ws_size >= ~310 MB).
// =====================================================================

#define BS_  2
#define N_   128
#define D_   512
#define H_   8
#define DH_  64
#define HID_ 2048

#define NH_     (BS_ * N_ * D_)            // 131072
#define ROWS_E  (BS_ * N_ * N_)            // 32768
#define NE_     ((long long)ROWS_E * D_)   // 16777216
#define NSC_    (BS_ * H_ * N_ * N_)       // 262144
#define SROWS_  (BS_ * H_ * N_)            // 2048

typedef __attribute__((ext_vector_type(16))) __bf16       v16bf;
typedef __attribute__((ext_vector_type(8)))  float        v8f;
typedef __attribute__((ext_vector_type(4)))  unsigned int u32x4;
typedef __attribute__((ext_vector_type(2)))  unsigned int u32x2;
typedef __attribute__((ext_vector_type(8)))  int          i32x8;
typedef __attribute__((ext_vector_type(4)))  int          i32x4;

#if defined(__has_builtin)
#  if __has_builtin(__builtin_amdgcn_tensor_load_to_lds) && \
      __has_builtin(__builtin_amdgcn_s_wait_tensorcnt)
#    define HAVE_TDM 1
#  else
#    define HAVE_TDM 0
#  endif
#else
#  define HAVE_TDM 0
#endif

union FragBF {
  struct { u32x4 lo, hi; } u;
  v16bf v;
};

// ---- fp32 <-> bf16 helpers (RNE) ------------------------------------
static __device__ __forceinline__ unsigned short f2bf(float f) {
  unsigned int u = __float_as_uint(f);
  u += 0x7FFFu + ((u >> 16) & 1u);
  return (unsigned short)(u >> 16);
}
static __device__ __forceinline__ float bf2f(unsigned int us) {
  return __uint_as_float(us << 16);
}
static __device__ __forceinline__ unsigned short loadcvt(const float* p) { return f2bf(*p); }
static __device__ __forceinline__ unsigned short loadcvt(const unsigned short* p) { return *p; }
static __device__ __forceinline__ void storecvt(float* p, float v) { *p = v; }
static __device__ __forceinline__ void storecvt(unsigned short* p, float v) { *p = f2bf(v); }

// pack 4 fp32 -> 4 bf16, one 8-byte LDS store
static __device__ __forceinline__ void store4bf(unsigned short* p, float4 v) {
  u32x2 pk;
  pk[0] = ((unsigned)f2bf(v.y) << 16) | (unsigned)f2bf(v.x);
  pk[1] = ((unsigned)f2bf(v.w) << 16) | (unsigned)f2bf(v.z);
  *(u32x2*)p = pk;
}

static __device__ __forceinline__ v8f wmma_bf16(v16bf a, v16bf b, v8f c) {
  return __builtin_amdgcn_wmma_f32_16x16x32_bf16(false, a, false, b, (short)0, c,
                                                 false, false);
}

#define BM 64
#define BN 64
#define BK 32
#define BKP 40   // padded LDS pitch (ushorts): 80 B per 64 B row (+16 B pad)

#if HAVE_TDM
// ---------------------------------------------------------------------
// TDM: DMA a [tileY x tileX] bf16 tile (row stride strideElems) from
// global memory into LDS at ldsOff, padding +4 DWORDs every 16 DWORDs
// so the LDS pitch becomes BKP(40) ushorts. D# per ISA 08_async_tensor:
//   g0: count=1 | lds_addr[63:32] | global_addr[120:64] | type=2[127:126]
//   g1: data_size=2B, pad_enable, pad_interval=16DW, pad_amount=4DW,
//       tensor_dim0/1, tile_dim0/1, tensor_dim0_stride
// Groups 2/3 unused for a 2D tile (zero-filled). 6-arg builtin form
// (clang-23 / therock-10.0 headers): (g0, g1, g2, g3, g4, cpol).
// ---------------------------------------------------------------------
static __device__ __forceinline__ void tdm_load_2d_bf16(
    unsigned ldsOff, const void* gaddr, unsigned tileX, unsigned tileY,
    unsigned long long strideElems, unsigned tensX, unsigned tensY) {
  unsigned long long ga = (unsigned long long)gaddr;
  u32x4 g0;
  g0[0] = 1u;  // count=1 (valid), user mode, no gather
  g0[1] = ldsOff;
  g0[2] = (unsigned)(ga & 0xffffffffu);
  g0[3] = (unsigned)((ga >> 32) & 0x01ffffffu) | (2u << 30);  // type=2
  i32x8 g1;
  g1[0] = (int)((1u << 16) | (1u << 20) | (3u << 22) | (3u << 25));
  g1[1] = (int)((tensX & 0xffffu) << 16);                       // dim0 lo16
  g1[2] = (int)(((tensX >> 16) & 0xffffu) | ((tensY & 0xffffu) << 16));
  g1[3] = (int)(((tensY >> 16) & 0xffffu) | ((tileX & 0xffffu) << 16));
  g1[4] = (int)(tileY & 0xffffu);                               // tile_dim1
  g1[5] = (int)(unsigned)(strideElems & 0xffffffffu);
  g1[6] = (int)(unsigned)((strideElems >> 32) & 0xffffu);
  g1[7] = 0;
  i32x4 z4 = {0, 0, 0, 0};
  i32x8 z8 = {0, 0, 0, 0, 0, 0, 0, 0};
  __builtin_amdgcn_tensor_load_to_lds(g0, g1, z4, z4, z8, 0);
}
#endif

// =====================================================================
// Generic WMMA GEMM:  C[M,N] = A[M,K] @ B[K,N]  (+ epilogue)
//   EPI 0 = none ; 1 = +bias+res ; 2 = +res ; 3 = +ni[i]+nj[j]
// Batched via blockIdx.z:  off = (z/div)*s1 + (z%div)*s2   per A/B/C.
// All shapes here: M%64==0, N%64==0, K%32==0.
// =====================================================================
template <int EPI, typename AT, typename OT>
__global__ __launch_bounds__(256) void gemm_k(
    const AT* __restrict__ A, const float* __restrict__ B, OT* __restrict__ C,
    int M, int N, int K, int lda, int ldb, int ldc,
    long long sA1, long long sA2, int divA,
    long long sB1, long long sB2, int divB,
    long long sC1, long long sC2, int divC,
    const float* __restrict__ bias, const float* __restrict__ res,
    const float* __restrict__ ni, const float* __restrict__ nj) {
  __shared__ unsigned short As[BM * BKP];
  __shared__ unsigned short Bs[BN * BKP];

  const int t    = threadIdx.x;
  const int lane = t & 31;
  const int w    = t >> 5;      // 8 wave32s
  const int wm   = w & 3;       // 4 waves along M
  const int wn   = w >> 2;      // 2 waves along N (16x32 per wave)

  const int rowBase = blockIdx.y * BM;
  const int colBase = blockIdx.x * BN;
  const int z       = blockIdx.z;
  const size_t aOff = (size_t)(z / divA) * sA1 + (size_t)(z % divA) * sA2;
  const size_t bOff = (size_t)(z / divB) * sB1 + (size_t)(z % divB) * sB2;
  const size_t cOff = (size_t)(z / divC) * sC1 + (size_t)(z % divC) * sC2;

  // ---- loop-invariant staging pointers (advance by BK per K step) ----
  // fp32 A: two float4 per thread (rows t>>3 and +32, cols (t&7)*4)
  const AT* aP0 = A + aOff + (size_t)(rowBase + (t >> 3)) * lda + (t & 7) * 4;
  const AT* aP1 = aP0 + (size_t)32 * lda;
  unsigned short* as0 = &As[(t >> 3) * BKP + (t & 7) * 4];
  unsigned short* as1 = as0 + 32 * BKP;
  // bf16 A (manual fallback): one u32x4 (8 elems) per thread
  const AT* aPb = A + aOff + (size_t)(rowBase + (t >> 2)) * lda + (t & 3) * 8;
  unsigned short* asb = &As[(t >> 2) * BKP + (t & 3) * 8];
  // TDM tile base
  const AT* aTd = A + aOff + (size_t)rowBase * lda;
  const unsigned asBase = (unsigned)(size_t)(void*)&As[0];
  (void)asBase; (void)aTd; (void)aPb; (void)asb;
  // B: two float4 per thread (k rows t>>4 and +16, cols (t&15)*4), store
  // transposed into Bs[n][k]
  const float* bP0 = B + bOff + (size_t)(t >> 4) * ldb + colBase + (t & 15) * 4;
  const float* bP1 = bP0 + (size_t)16 * ldb;
  const int bkk = t >> 4, bnn = (t & 15) * 4;

  v8f acc0 = {0.f, 0.f, 0.f, 0.f, 0.f, 0.f, 0.f, 0.f};
  v8f acc1 = acc0;

  const int lm = lane & 15, lh = lane >> 4;
  const unsigned short* ap  = &As[(wm * 16 + lm) * BKP + lh * 8];
  const unsigned short* bp0 = &Bs[(wn * 32 + lm) * BKP + lh * 8];
  const unsigned short* bp1 = bp0 + 16 * BKP;

  for (int kt = 0; kt < K; kt += BK) {
    __syncthreads();
    // ---- stage A tile (64x32 -> bf16 LDS) ----
    if constexpr (sizeof(AT) == 2) {
#if HAVE_TDM
      if (w == 0)
        tdm_load_2d_bf16(asBase, (const void*)(aTd + kt), BK, BM,
                         (unsigned long long)lda, (unsigned)lda, (unsigned)M);
#else
      *(u32x4*)asb = *(const u32x4*)(aPb + kt);
#endif
    } else {
      float4 va0 = *(const float4*)(aP0 + kt);
      float4 va1 = *(const float4*)(aP1 + kt);
      store4bf(as0, va0);
      store4bf(as1, va1);
    }
    // ---- stage B tile (32x64), transposed -> Bs[n][k] ----
    {
      float4 vb0 = *(const float4*)bP0;
      float4 vb1 = *(const float4*)bP1;
      Bs[(bnn + 0) * BKP + bkk] = f2bf(vb0.x);
      Bs[(bnn + 1) * BKP + bkk] = f2bf(vb0.y);
      Bs[(bnn + 2) * BKP + bkk] = f2bf(vb0.z);
      Bs[(bnn + 3) * BKP + bkk] = f2bf(vb0.w);
      Bs[(bnn + 0) * BKP + bkk + 16] = f2bf(vb1.x);
      Bs[(bnn + 1) * BKP + bkk + 16] = f2bf(vb1.y);
      Bs[(bnn + 2) * BKP + bkk + 16] = f2bf(vb1.z);
      Bs[(bnn + 3) * BKP + bkk + 16] = f2bf(vb1.w);
      bP0 += (size_t)BK * ldb;
      bP1 += (size_t)BK * ldb;
    }
    if (kt + BK < K) {  // global_prefetch_b8 on next K tile
      __builtin_prefetch(aP0 + kt + BK, 0, 0);
      __builtin_prefetch(bP0, 0, 0);
    }
#if HAVE_TDM
    if constexpr (sizeof(AT) == 2) {
      if (w == 0) __builtin_amdgcn_s_wait_tensorcnt(0);
    }
#endif
    __syncthreads();

    FragBF fa, fb0, fb1;
    fa.u.lo  = *(const u32x4*)ap;
    fa.u.hi  = *(const u32x4*)(ap + 16);
    fb0.u.lo = *(const u32x4*)bp0;
    fb0.u.hi = *(const u32x4*)(bp0 + 16);
    fb1.u.lo = *(const u32x4*)bp1;
    fb1.u.hi = *(const u32x4*)(bp1 + 16);
    acc0 = wmma_bf16(fa.v, fb0.v, acc0);
    acc1 = wmma_bf16(fa.v, fb1.v, acc1);
  }

  v8f accs[2] = {acc0, acc1};
#pragma unroll
  for (int s = 0; s < 2; ++s) {
#pragma unroll
    for (int r = 0; r < 8; ++r) {
      int row = rowBase + wm * 16 + lh * 8 + r;
      int col = colBase + wn * 32 + s * 16 + lm;
      if (row >= M || col >= N) continue;
      float v = accs[s][r];
      if (EPI == 1) {
        v += bias[col] + res[cOff + (size_t)row * ldc + col];
      } else if (EPI == 2) {
        v += res[cOff + (size_t)row * ldc + col];
      } else if (EPI == 3) {
        int b = row >> 14, ij = row & 16383, ii = ij >> 7, jj = ij & 127;
        v += ni[(size_t)((b << 7) + ii) * D_ + col] +
             nj[(size_t)((b << 7) + jj) * D_ + col];
      }
      storecvt(C + cOff + (size_t)row * ldc + col, v);
    }
  }
}

// =====================================================================
// Fused dual GEMM SwiGLU:  C = silu(A@B1) * (A@B3)
// =====================================================================
template <typename AT, typename OT>
__global__ __launch_bounds__(256) void gemm_swiglu_k(
    const AT* __restrict__ A, const float* __restrict__ B1,
    const float* __restrict__ B3, OT* __restrict__ C,
    int M, int N, int K, int lda, int ldb, int ldc) {
  __shared__ unsigned short As[BM * BKP];
  __shared__ unsigned short B1s[BN * BKP];
  __shared__ unsigned short B3s[BN * BKP];

  const int t    = threadIdx.x;
  const int lane = t & 31;
  const int w    = t >> 5;
  const int wm   = w & 3;
  const int wn   = w >> 2;
  const int rowBase = blockIdx.y * BM;
  const int colBase = blockIdx.x * BN;

  const AT* aP0 = A + (size_t)(rowBase + (t >> 3)) * lda + (t & 7) * 4;
  const AT* aP1 = aP0 + (size_t)32 * lda;
  unsigned short* as0 = &As[(t >> 3) * BKP + (t & 7) * 4];
  unsigned short* as1 = as0 + 32 * BKP;
  const AT* aPb = A + (size_t)(rowBase + (t >> 2)) * lda + (t & 3) * 8;
  unsigned short* asb = &As[(t >> 2) * BKP + (t & 3) * 8];
  const AT* aTd = A + (size_t)rowBase * lda;
  const unsigned asBase = (unsigned)(size_t)(void*)&As[0];
  (void)asBase; (void)aTd; (void)aPb; (void)asb;
  const float* b1P0 = B1 + (size_t)(t >> 4) * ldb + colBase + (t & 15) * 4;
  const float* b1P1 = b1P0 + (size_t)16 * ldb;
  const float* b3P0 = B3 + (size_t)(t >> 4) * ldb + colBase + (t & 15) * 4;
  const float* b3P1 = b3P0 + (size_t)16 * ldb;
  const int bkk = t >> 4, bnn = (t & 15) * 4;

  v8f g0 = {0.f, 0.f, 0.f, 0.f, 0.f, 0.f, 0.f, 0.f};
  v8f g1 = g0, x0 = g0, x1 = g0;

  const int lm = lane & 15, lh = lane >> 4;
  const unsigned short* ap   = &As[(wm * 16 + lm) * BKP + lh * 8];
  const unsigned short* q10  = &B1s[(wn * 32 + lm) * BKP + lh * 8];
  const unsigned short* q11  = q10 + 16 * BKP;
  const unsigned short* q30  = &B3s[(wn * 32 + lm) * BKP + lh * 8];
  const unsigned short* q31  = q30 + 16 * BKP;

  for (int kt = 0; kt < K; kt += BK) {
    __syncthreads();
    if constexpr (sizeof(AT) == 2) {
#if HAVE_TDM
      if (w == 0)
        tdm_load_2d_bf16(asBase, (const void*)(aTd + kt), BK, BM,
                         (unsigned long long)lda, (unsigned)lda, (unsigned)M);
#else
      *(u32x4*)asb = *(const u32x4*)(aPb + kt);
#endif
    } else {
      float4 va0 = *(const float4*)(aP0 + kt);
      float4 va1 = *(const float4*)(aP1 + kt);
      store4bf(as0, va0);
      store4bf(as1, va1);
    }
    {
      float4 v10 = *(const float4*)b1P0;
      float4 v11 = *(const float4*)b1P1;
      float4 v30 = *(const float4*)b3P0;
      float4 v31 = *(const float4*)b3P1;
      B1s[(bnn + 0) * BKP + bkk] = f2bf(v10.x);
      B1s[(bnn + 1) * BKP + bkk] = f2bf(v10.y);
      B1s[(bnn + 2) * BKP + bkk] = f2bf(v10.z);
      B1s[(bnn + 3) * BKP + bkk] = f2bf(v10.w);
      B1s[(bnn + 0) * BKP + bkk + 16] = f2bf(v11.x);
      B1s[(bnn + 1) * BKP + bkk + 16] = f2bf(v11.y);
      B1s[(bnn + 2) * BKP + bkk + 16] = f2bf(v11.z);
      B1s[(bnn + 3) * BKP + bkk + 16] = f2bf(v11.w);
      B3s[(bnn + 0) * BKP + bkk] = f2bf(v30.x);
      B3s[(bnn + 1) * BKP + bkk] = f2bf(v30.y);
      B3s[(bnn + 2) * BKP + bkk] = f2bf(v30.z);
      B3s[(bnn + 3) * BKP + bkk] = f2bf(v30.w);
      B3s[(bnn + 0) * BKP + bkk + 16] = f2bf(v31.x);
      B3s[(bnn + 1) * BKP + bkk + 16] = f2bf(v31.y);
      B3s[(bnn + 2) * BKP + bkk + 16] = f2bf(v31.z);
      B3s[(bnn + 3) * BKP + bkk + 16] = f2bf(v31.w);
      b1P0 += (size_t)BK * ldb; b1P1 += (size_t)BK * ldb;
      b3P0 += (size_t)BK * ldb; b3P1 += (size_t)BK * ldb;
    }
    if (kt + BK < K) {
      __builtin_prefetch(aP0 + kt + BK, 0, 0);
      __builtin_prefetch(b1P0, 0, 0);
      __builtin_prefetch(b3P0, 0, 0);
    }
#if HAVE_TDM
    if constexpr (sizeof(AT) == 2) {
      if (w == 0) __builtin_amdgcn_s_wait_tensorcnt(0);
    }
#endif
    __syncthreads();

    FragBF fa, f10, f11, f30, f31;
    fa.u.lo  = *(const u32x4*)ap;  fa.u.hi  = *(const u32x4*)(ap + 16);
    f10.u.lo = *(const u32x4*)q10; f10.u.hi = *(const u32x4*)(q10 + 16);
    f11.u.lo = *(const u32x4*)q11; f11.u.hi = *(const u32x4*)(q11 + 16);
    f30.u.lo = *(const u32x4*)q30; f30.u.hi = *(const u32x4*)(q30 + 16);
    f31.u.lo = *(const u32x4*)q31; f31.u.hi = *(const u32x4*)(q31 + 16);
    g0 = wmma_bf16(fa.v, f10.v, g0);
    g1 = wmma_bf16(fa.v, f11.v, g1);
    x0 = wmma_bf16(fa.v, f30.v, x0);
    x1 = wmma_bf16(fa.v, f31.v, x1);
  }

  v8f gs[2] = {g0, g1}, xs[2] = {x0, x1};
#pragma unroll
  for (int s = 0; s < 2; ++s) {
#pragma unroll
    for (int r = 0; r < 8; ++r) {
      int row = rowBase + wm * 16 + lh * 8 + r;
      int col = colBase + wn * 32 + s * 16 + lm;
      if (row >= M || col >= N) continue;
      float g = gs[s][r];
      float u = g * (1.f / (1.f + __expf(-g)));  // silu
      storecvt(C + (size_t)row * ldc + col, u * xs[s][r]);
    }
  }
}

// ---- BatchNorm1d over (bs,n) per channel; thread-per-channel --------
__global__ __launch_bounds__(256) void bn_k(const float* __restrict__ x,
                                            const float* __restrict__ g,
                                            const float* __restrict__ b,
                                            float* __restrict__ y) {
  int c = blockIdx.x * 256 + threadIdx.x;
  float s = 0.f, q = 0.f;
  for (int r = 0; r < BS_ * N_; ++r) {
    float v = x[(size_t)r * D_ + c];
    s += v;
    q += v * v;
  }
  float m   = s * (1.f / (BS_ * N_));
  float var = q * (1.f / (BS_ * N_)) - m * m;
  float sc  = rsqrtf(var + 1e-5f) * g[c];
  float sh  = b[c] - m * sc;
  for (int r = 0; r < BS_ * N_; ++r)
    y[(size_t)r * D_ + c] = x[(size_t)r * D_ + c] * sc + sh;
}

// ---- LayerNorm over last dim (512); one wave32 per row --------------
template <typename OT>
__global__ __launch_bounds__(256) void ln_k(const float* __restrict__ x,
                                            const float* __restrict__ g,
                                            const float* __restrict__ b,
                                            OT* __restrict__ y, int rows) {
  int gw   = (blockIdx.x * 256 + threadIdx.x) >> 5;
  int lane = threadIdx.x & 31;
  if (gw >= rows) return;
  const float* xr = x + (size_t)gw * D_;
  float4 v[4];
  float s = 0.f;
#pragma unroll
  for (int c = 0; c < 4; ++c) {
    v[c] = *(const float4*)&xr[c * 128 + lane * 4];
    s += v[c].x + v[c].y + v[c].z + v[c].w;
  }
  for (int m = 16; m; m >>= 1) s += __shfl_xor(s, m, 32);
  float mean = s * (1.f / D_);
  float q = 0.f;
#pragma unroll
  for (int c = 0; c < 4; ++c) {
    float dx = v[c].x - mean, dy = v[c].y - mean, dz = v[c].z - mean, dw = v[c].w - mean;
    q += dx * dx + dy * dy + dz * dz + dw * dw;
  }
  for (int m = 16; m; m >>= 1) q += __shfl_xor(q, m, 32);
  float inv = rsqrtf(q * (1.f / D_) + 1e-5f);
#pragma unroll
  for (int c = 0; c < 4; ++c) {
    int cc = c * 128 + lane * 4;
    storecvt(&y[(size_t)gw * D_ + cc + 0], (v[c].x - mean) * inv * g[cc + 0] + b[cc + 0]);
    storecvt(&y[(size_t)gw * D_ + cc + 1], (v[c].y - mean) * inv * g[cc + 1] + b[cc + 1]);
    storecvt(&y[(size_t)gw * D_ + cc + 2], (v[c].z - mean) * inv * g[cc + 2] + b[cc + 2]);
    storecvt(&y[(size_t)gw * D_ + cc + 3], (v[c].w - mean) * inv * g[cc + 3] + b[cc + 3]);
  }
}

// ---- scores = q.(ev o k)/8; one wave32 per (b,h,i,j) ----------------
__global__ __launch_bounds__(256) void scores_k(const float* __restrict__ q,
                                                const float* __restrict__ k,
                                                const unsigned short* __restrict__ ev,
                                                float* __restrict__ s) {
  int gw   = (blockIdx.x * 256 + threadIdx.x) >> 5;
  int lane = threadIdx.x & 31;
  int j = gw & 127, i = (gw >> 7) & 127, h = (gw >> 14) & 7, b = gw >> 17;
  float2 qv = *(const float2*)&q[(size_t)(b * N_ + i) * D_ + h * DH_ + lane * 2];
  float2 kv = *(const float2*)&k[(size_t)(b * N_ + j) * D_ + h * DH_ + lane * 2];
  unsigned int ep =
      *(const unsigned int*)&ev[((size_t)(b * N_ + i) * N_ + j) * D_ + h * DH_ + lane * 2];
  float p = qv.x * bf2f(ep & 0xffffu) * kv.x + qv.y * bf2f(ep >> 16) * kv.y;
  for (int m = 16; m; m >>= 1) p += __shfl_xor(p, m, 32);
  if (lane == 0) s[gw] = p * 0.125f;
}

// ---- softmax over j (128), one wave per (b,h,i), in place -----------
__global__ __launch_bounds__(256) void softmax_k(float* __restrict__ s) {
  int gw   = (blockIdx.x * 256 + threadIdx.x) >> 5;
  int lane = threadIdx.x & 31;
  float* r = s + (size_t)gw * N_;
  float4 x = *(const float4*)&r[lane * 4];
  float m = fmaxf(fmaxf(x.x, x.y), fmaxf(x.z, x.w));
  for (int k = 16; k; k >>= 1) m = fmaxf(m, __shfl_xor(m, k, 32));
  x.x = __expf(x.x - m); x.y = __expf(x.y - m);
  x.z = __expf(x.z - m); x.w = __expf(x.w - m);
  float sum = x.x + x.y + x.z + x.w;
  for (int k = 16; k; k >>= 1) sum += __shfl_xor(sum, k, 32);
  float inv = 1.f / sum;
  x.x *= inv; x.y *= inv; x.z *= inv; x.w *= inv;
  *(float4*)&r[lane * 4] = x;
}

// ---- in-place sym_tensor (writes only strict lower triangle) --------
__global__ __launch_bounds__(256) void sym_k(float* __restrict__ e) {
  int idx = blockIdx.x * 256 + threadIdx.x;
  int d = idx & 511, j = (idx >> 9) & 127, i = (idx >> 16) & 127;
  if (i <= j) return;
  int b = idx >> 23;
  float up = e[((size_t)((b << 7) + j) * N_ + i) * D_ + d];
  if (fabsf(up) > 0.f) e[idx] = up;
}

// =====================================================================
extern "C" void kernel_launch(void* const* d_in, const int* in_sizes, int n_in,
                              void* d_out, int out_size, void* d_ws, size_t ws_size,
                              hipStream_t stream) {
  const float* h     = (const float*)d_in[0];
  const float* e     = (const float*)d_in[1];
  const float* Wq    = (const float*)d_in[2];
  const float* Wk    = (const float*)d_in[3];
  const float* Wv    = (const float*)d_in[4];
  const float* We    = (const float*)d_in[5];
  const float* Wni   = (const float*)d_in[6];
  const float* Wnj   = (const float*)d_in[7];
  const float* Woh   = (const float*)d_in[8];
  const float* boh   = (const float*)d_in[9];
  const float* Woe   = (const float*)d_in[10];
  const float* boe   = (const float*)d_in[11];
  const float* bn1_g = (const float*)d_in[12];
  const float* bn1_b = (const float*)d_in[13];
  const float* bn2_g = (const float*)d_in[14];
  const float* bn2_b = (const float*)d_in[15];
  const float* ln1_g = (const float*)d_in[16];
  const float* ln1_b = (const float*)d_in[17];
  const float* ln2_g = (const float*)d_in[18];
  const float* ln2_b = (const float*)d_in[19];
  const float* fh1   = (const float*)d_in[20];
  const float* fh3   = (const float*)d_in[21];
  const float* fh2   = (const float*)d_in[22];
  const float* fe1   = (const float*)d_in[23];
  const float* fe3   = (const float*)d_in[24];
  const float* fe2   = (const float*)d_in[25];

  char* ws = (char*)d_ws;
  size_t off = 0;
  float* e1 = (float*)(ws + off);                     off += (size_t)NE_ * 4;
  unsigned short* evbf = (unsigned short*)(ws + off); off += (size_t)NE_ * 2;
  unsigned short* e2n  = evbf;  // reuse: ev dead after ev@Woe
  float* e2 = (float*)(ws + off);                     off += (size_t)NE_ * 4;
  unsigned short* ue = (unsigned short*)(ws + off);   off += (size_t)ROWS_E * HID_ * 2;
  float* sc = (float*)(ws + off);                     off += (size_t)NSC_ * 4;
  float* h1   = (float*)(ws + off); off += (size_t)NH_ * 4;
  float* qb   = (float*)(ws + off); off += (size_t)NH_ * 4;
  float* kb   = (float*)(ws + off); off += (size_t)NH_ * 4;
  float* vb   = (float*)(ws + off); off += (size_t)NH_ * 4;
  float* nib  = (float*)(ws + off); off += (size_t)NH_ * 4;
  float* njb  = (float*)(ws + off); off += (size_t)NH_ * 4;
  float* hain = (float*)(ws + off); off += (size_t)NH_ * 4;
  float* h2   = (float*)(ws + off); off += (size_t)NH_ * 4;
  float* h2n  = (float*)(ws + off); off += (size_t)NH_ * 4;
  float* uh   = (float*)(ws + off); off += (size_t)(BS_ * N_) * HID_ * 4;

  float* outh = (float*)d_out;
  float* oute = outh + NH_;

  bn_k<<<dim3(2), dim3(256), 0, stream>>>(h, bn1_g, bn1_b, h1);
  ln_k<float><<<dim3(4096), dim3(256), 0, stream>>>(e, ln1_g, ln1_b, e1, ROWS_E);

  const float* Wl[5] = {Wq, Wk, Wv, Wni, Wnj};
  float* Ol[5] = {qb, kb, vb, nib, njb};
  for (int p = 0; p < 5; ++p) {
    gemm_k<0, float, float><<<dim3(8, 4, 1), dim3(256), 0, stream>>>(
        h1, Wl[p], Ol[p], 256, 512, 512, 512, 512, 512,
        0, 0, 1, 0, 0, 1, 0, 0, 1, nullptr, nullptr, nullptr, nullptr);
  }

  gemm_k<3, float, unsigned short><<<dim3(8, 512, 1), dim3(256), 0, stream>>>(
      e1, We, evbf, ROWS_E, 512, 512, 512, 512, 512,
      0, 0, 1, 0, 0, 1, 0, 0, 1, nullptr, nullptr, nib, njb);

  scores_k<<<dim3(NSC_ / 8), dim3(256), 0, stream>>>(qb, kb, evbf, sc);
  softmax_k<<<dim3(SROWS_ / 8), dim3(256), 0, stream>>>(sc);

  gemm_k<0, float, float><<<dim3(1, 2, BS_ * H_), dim3(256), 0, stream>>>(
      sc, vb, hain, 128, 64, 128, 128, 512, 512,
      (long long)N_ * N_, 0, 1,
      (long long)N_ * D_, DH_, H_,
      (long long)N_ * D_, DH_, H_,
      nullptr, nullptr, nullptr, nullptr);

  gemm_k<1, float, float><<<dim3(8, 4, 1), dim3(256), 0, stream>>>(
      hain, Woh, h2, 256, 512, 512, 512, 512, 512,
      0, 0, 1, 0, 0, 1, 0, 0, 1, boh, h1, nullptr, nullptr);

  gemm_k<1, unsigned short, float><<<dim3(8, 512, 1), dim3(256), 0, stream>>>(
      evbf, Woe, e2, ROWS_E, 512, 512, 512, 512, 512,
      0, 0, 1, 0, 0, 1, 0, 0, 1, boe, e1, nullptr, nullptr);

  bn_k<<<dim3(2), dim3(256), 0, stream>>>(h2, bn2_g, bn2_b, h2n);
  ln_k<unsigned short><<<dim3(4096), dim3(256), 0, stream>>>(e2, ln2_g, ln2_b, e2n, ROWS_E);

  gemm_swiglu_k<float, float><<<dim3(32, 4, 1), dim3(256), 0, stream>>>(
      h2n, fh1, fh3, uh, 256, HID_, 512, 512, HID_, HID_);

  gemm_k<2, float, float><<<dim3(8, 4, 1), dim3(256), 0, stream>>>(
      uh, fh2, outh, 256, 512, HID_, HID_, 512, 512,
      0, 0, 1, 0, 0, 1, 0, 0, 1, nullptr, h2, nullptr, nullptr);

  gemm_swiglu_k<unsigned short, unsigned short><<<dim3(32, 512, 1), dim3(256), 0, stream>>>(
      e2n, fe1, fe3, ue, ROWS_E, HID_, 512, 512, HID_, HID_);

  gemm_k<2, unsigned short, float><<<dim3(8, 512, 1), dim3(256), 0, stream>>>(
      ue, fe2, oute, ROWS_E, 512, HID_, HID_, 512, 512,
      0, 0, 1, 0, 0, 1, 0, 0, 1, nullptr, e2, nullptr, nullptr);

  sym_k<<<dim3((unsigned)(NE_ / 256)), dim3(256), 0, stream>>>(oute);
}
